// ECCNN_32633161515376
// MI455X (gfx1250) — matrix operand order repaired and checked by reading, not compiled, not run on hardware
//
#include <hip/hip_runtime.h>
#include <stdint.h>

#define NN   32768
#define EE   98304
#define DD   64
#define GG   64
#define OUTD 16
#define W2KS 130   // 4096/32 k-steps for W2 + 2 k-steps for b2 tail
#define BF1DUP 0x3F803F80u

typedef __attribute__((ext_vector_type(16))) __bf16 v16bf;
typedef __attribute__((ext_vector_type(8)))  float  v8f;

struct __align__(16) Q4 { unsigned x, y, z, w; };
union AB { Q4 q[2]; unsigned u[8]; v16bf v; };

__device__ __forceinline__ unsigned short f2bf(float f) {
  unsigned u = __float_as_uint(f);
  return (unsigned short)((u + 0x7FFFu + ((u >> 16) & 1u)) >> 16);  // RNE
}

__device__ __forceinline__ unsigned pk_mul_bf16(unsigned a, unsigned b) {
  unsigned d;
  asm("v_pk_mul_bf16 %0, %1, %2" : "=v"(d) : "v"(a), "v"(b));
  return d;
}

// ---------------- small utility kernels ----------------

__global__ void zero_f32(float* p, int n) {
  int i = blockIdx.x * blockDim.x + threadIdx.x;
  if (i < n) p[i] = 0.f;
}

// f32 [rows, icols] -> bf16 [rows, ocols], zero-padded cols
__global__ void cvt_bf16(const float* in, unsigned short* out, int rows, int icols, int ocols) {
  int i = blockIdx.x * blockDim.x + threadIdx.x;
  if (i >= rows * ocols) return;
  int r = i / ocols, c = i - r * ocols;
  out[i] = (c < icols) ? f2bf(in[(size_t)r * icols + c]) : (unsigned short)0;
}

// Pack row-major W [K_actual, 64] into WMMA B-operand lane order (bf16),
// zero-filling up to ksteps*32 rows. Layout: block per (kstep,ntile):
// 32 lanes x 16 bf16; element e of lane L is B[kstep*32 + (L<16?0:8)+(e&7)+(e>>3)*16, ntile*16 + (L&15)].
__global__ void pack_w(const float* W, int K_actual, int ksteps, unsigned short* out) {
  int idx = blockIdx.x * blockDim.x + threadIdx.x;
  int total = ksteps * 4 * 32 * 16;
  if (idx >= total) return;
  int e = idx & 15, lane = (idx >> 4) & 31, nt = (idx >> 9) & 3, ks = idx >> 11;
  int kloc = ((lane < 16) ? 0 : 8) + (e & 7) + ((e >> 3) << 4);
  int k = ks * 32 + kloc;
  int o = nt * 16 + (lane & 15);
  float v = (k < K_actual) ? W[(size_t)k * 64 + o] : 0.f;
  out[idx] = f2bf(v);
}

// Pack W2r[(d,i), o] = W2[d, i*64+o]  (W2 is [64,4096]), plus b2 tail rows:
// rows 4096..4159 are B2r[i,o] = b2[i*64+o]. Same B lane layout, 130 ksteps.
__global__ void pack_w2(const float* W2, const float* b2, unsigned short* out) {
  int idx = blockIdx.x * blockDim.x + threadIdx.x;
  int total = W2KS * 4 * 32 * 16;
  if (idx >= total) return;
  int e = idx & 15, lane = (idx >> 4) & 31, nt = (idx >> 9) & 3, ks = idx >> 11;
  int kloc = ((lane < 16) ? 0 : 8) + (e & 7) + ((e >> 3) << 4);
  int k = ks * 32 + kloc;
  int o = nt * 16 + (lane & 15);
  float v;
  if (k < 4096) v = W2[(size_t)(k >> 6) * 4096 + (size_t)(k & 63) * 64 + o];
  else          v = b2[(size_t)(k - 4096) * 64 + o];
  out[idx] = f2bf(v);
}

// ---------------- generic WMMA GEMM: C[M,64] = A[M,K] @ Bpacked + bias ----------------
// One wave per 16-row tile of A. A is bf16 row-major with row stride lda (K<=lda).
__global__ void gemm16(const unsigned short* __restrict__ Abf, int lda, int Mtiles, int ksteps,
                       const unsigned short* __restrict__ Bp, const float* __restrict__ bias,
                       float* outf, unsigned short* outbf, unsigned* outdup, int relu) {
  int wid  = (int)((blockIdx.x * blockDim.x + threadIdx.x) >> 5);
  int lane = threadIdx.x & 31;
  if (wid >= Mtiles) return;                 // wave-uniform: EXEC stays all-1s
  int row = lane & 15, half = lane >> 4;
  const unsigned short* arow = Abf + (size_t)(wid * 16 + row) * lda;
  int baseK = half * 8;

  v8f zero = {0.f, 0.f, 0.f, 0.f, 0.f, 0.f, 0.f, 0.f};
  v8f c[4] = {zero, zero, zero, zero};

  for (int ks = 0; ks < ksteps; ++ks) {
    AB a;
    a.q[0] = *(const Q4*)(arow + ks * 32 + baseK);        // K = baseK+0..7
    a.q[1] = *(const Q4*)(arow + ks * 32 + 16 + baseK);   // K = 16+baseK+0..7
#pragma unroll
    for (int t = 0; t < 4; ++t) {
      const unsigned short* bp = Bp + ((size_t)(ks * 4 + t) * 32 + lane) * 16;
      AB b;
      b.q[0] = *(const Q4*)bp;
      b.q[1] = *(const Q4*)(bp + 8);
      c[t] = __builtin_amdgcn_wmma_f32_16x16x32_bf16(false, a.v, false, b.v,
                                                     (short)0, c[t], false, false);
    }
  }

  int col0 = lane & 15;
#pragma unroll
  for (int t = 0; t < 4; ++t) {
    float bv = bias ? bias[t * 16 + col0] : 0.f;
#pragma unroll
    for (int v = 0; v < 8; ++v) {
      float val = c[t][v] + bv;
      if (relu) val = fmaxf(val, 0.f);
      int rm = wid * 16 + v + half * 8;
      size_t off = (size_t)rm * 64 + t * 16 + col0;
      if (outf)   outf[off] = val;
      if (outbf)  outbf[off] = f2bf(val);
      if (outdup) { unsigned s = f2bf(val); outdup[off] = (s << 16) | s; }
    }
  }
}

// ---------------- fused NNConv message GEMM ----------------
// msg[E,64] = K[E,4160] @ W2packed, where K[e, d*64+i] = H[e,d] * X[src_e, i],
// built on the fly with v_pk_mul_bf16; last 64 k-rows use scale 1.0 (b2 term).
// M=32 per wave (two 16-row tiles share every B fetch); per-lane X operand
// (32 bf16) hoisted into registers before the k-loop; H scale loaded once per d.
// Result scatter-added into agg[dst] with f32 atomics.
__global__ void nnconv_msg(const unsigned short* __restrict__ Xbf,  // [NN,64] bf16
                           const unsigned* __restrict__ Hdup,       // [EE,64] bf16 dup in u32
                           const unsigned short* __restrict__ Wp,   // packed, 130 ksteps
                           const int* __restrict__ src, const int* __restrict__ dst,
                           float* agg, int Ewtiles) {
  int wid  = (int)((blockIdx.x * blockDim.x + threadIdx.x) >> 5);
  int lane = threadIdx.x & 31;
  if (wid >= Ewtiles) return;                // wave-uniform exit
  int row = lane & 15, half = lane >> 4;
  int e0 = wid * 32 + row;
  int e1 = e0 + 16;
  int s0 = src[e0], s1 = src[e1];
  const unsigned short* xr0 = Xbf + (size_t)s0 * 64;
  const unsigned short* xr1 = Xbf + (size_t)s1 * 64;
  const unsigned* hr0 = Hdup + (size_t)e0 * 64;
  const unsigned* hr1 = Hdup + (size_t)e1 * 64;
  int baseK = half * 8;

  // Hoist this lane's X slices: quarters q cover elements q*16+baseK .. +7
  Q4 x0[4], x1[4];
#pragma unroll
  for (int q = 0; q < 4; ++q) {
    x0[q] = *(const Q4*)(xr0 + q * 16 + baseK);
    x1[q] = *(const Q4*)(xr1 + q * 16 + baseK);
  }

  v8f zero = {0.f, 0.f, 0.f, 0.f, 0.f, 0.f, 0.f, 0.f};
  v8f c0[4] = {zero, zero, zero, zero};
  v8f c1[4] = {zero, zero, zero, zero};

  for (int d = 0; d <= 64; ++d) {            // d==64 is the b2 tail (scale 1.0)
    unsigned sc0, sc1;
    if (d < 64) { sc0 = hr0[d]; sc1 = hr1[d]; }
    else        { sc0 = BF1DUP; sc1 = BF1DUP; }
#pragma unroll
    for (int p = 0; p < 2; ++p) {            // i0 = p*32 ; kstep = 2*d + p
      int ks = 2 * d + p;
      AB a0, a1;
      a0.u[0] = pk_mul_bf16(x0[2 * p].x, sc0);
      a0.u[1] = pk_mul_bf16(x0[2 * p].y, sc0);
      a0.u[2] = pk_mul_bf16(x0[2 * p].z, sc0);
      a0.u[3] = pk_mul_bf16(x0[2 * p].w, sc0);
      a0.u[4] = pk_mul_bf16(x0[2 * p + 1].x, sc0);
      a0.u[5] = pk_mul_bf16(x0[2 * p + 1].y, sc0);
      a0.u[6] = pk_mul_bf16(x0[2 * p + 1].z, sc0);
      a0.u[7] = pk_mul_bf16(x0[2 * p + 1].w, sc0);
      a1.u[0] = pk_mul_bf16(x1[2 * p].x, sc1);
      a1.u[1] = pk_mul_bf16(x1[2 * p].y, sc1);
      a1.u[2] = pk_mul_bf16(x1[2 * p].z, sc1);
      a1.u[3] = pk_mul_bf16(x1[2 * p].w, sc1);
      a1.u[4] = pk_mul_bf16(x1[2 * p + 1].x, sc1);
      a1.u[5] = pk_mul_bf16(x1[2 * p + 1].y, sc1);
      a1.u[6] = pk_mul_bf16(x1[2 * p + 1].z, sc1);
      a1.u[7] = pk_mul_bf16(x1[2 * p + 1].w, sc1);
#pragma unroll
      for (int t = 0; t < 4; ++t) {
        const unsigned short* bp = Wp + ((size_t)(ks * 4 + t) * 32 + lane) * 16;
        AB b;
        b.q[0] = *(const Q4*)bp;
        b.q[1] = *(const Q4*)(bp + 8);
        c0[t] = __builtin_amdgcn_wmma_f32_16x16x32_bf16(false, a0.v, false, b.v,
                                                        (short)0, c0[t], false, false);
        c1[t] = __builtin_amdgcn_wmma_f32_16x16x32_bf16(false, a1.v, false, b.v,
                                                        (short)0, c1[t], false, false);
      }
    }
  }

  int dn0[8], dn1[8];
#pragma unroll
  for (int v = 0; v < 8; ++v) {
    dn0[v] = dst[wid * 32 + v + half * 8];
    dn1[v] = dst[wid * 32 + 16 + v + half * 8];
  }
  int col0 = lane & 15;
#pragma unroll
  for (int t = 0; t < 4; ++t)
#pragma unroll
    for (int v = 0; v < 8; ++v) {
      atomicAdd(&agg[(size_t)dn0[v] * 64 + t * 16 + col0], c0[t][v]);
      atomicAdd(&agg[(size_t)dn1[v] * 64 + t * 16 + col0], c1[t][v]);
    }
}

// ---------------- layernorm / pooling / head ----------------

__global__ void stats_k(const float* __restrict__ x, size_t n, float* out2) {
  __shared__ float ssum[256], ssq[256];
  float s = 0.f, q = 0.f;
  for (size_t i = (size_t)blockIdx.x * blockDim.x + threadIdx.x; i < n;
       i += (size_t)gridDim.x * blockDim.x) {
    float v = x[i]; s += v; q += v * v;
  }
  ssum[threadIdx.x] = s; ssq[threadIdx.x] = q;
  __syncthreads();
  for (int k = 128; k > 0; k >>= 1) {
    if ((int)threadIdx.x < k) {
      ssum[threadIdx.x] += ssum[threadIdx.x + k];
      ssq[threadIdx.x]  += ssq[threadIdx.x + k];
    }
    __syncthreads();
  }
  if (threadIdx.x == 0) { atomicAdd(&out2[0], ssum[0]); atomicAdd(&out2[1], ssq[0]); }
}

__global__ void ln_relu(const float* __restrict__ a, const float* __restrict__ st,
                        const float* __restrict__ w, const float* __restrict__ b,
                        float* outf, unsigned short* outbf, int n) {
  int i = blockIdx.x * blockDim.x + threadIdx.x;
  if (i >= n) return;
  float inv = 1.f / (float)n;
  float mean = st[0] * inv;
  float var  = fmaxf(st[1] * inv - mean * mean, 0.f);
  float rs = 1.f / (sqrtf(var) + 1e-5f);
  int col = i & 63;
  float v = (a[i] - mean) * rs * w[col] + b[col];
  v = fmaxf(v, 0.f);
  outf[i] = v;
  if (outbf) outbf[i] = f2bf(v);
}

__global__ void pool_k(const float* __restrict__ h, const int* __restrict__ batch,
                       float* pool, float* cnt, int n) {
  int i = blockIdx.x * blockDim.x + threadIdx.x;
  if (i >= n * 64) return;
  int node = i >> 6, c = i & 63;
  int g = batch[node];
  atomicAdd(&pool[g * 64 + c], h[i]);
  if (c == 0) atomicAdd(&cnt[g], 1.f);
}

__global__ void head_k(const float* __restrict__ pool, const float* __restrict__ cnt,
                       const float* __restrict__ w, const float* __restrict__ b,
                       float* out) {
  int i = blockIdx.x * blockDim.x + threadIdx.x;
  if (i >= GG * OUTD) return;
  int g = i >> 4, o = i & 15;
  float cc = fmaxf(cnt[g], 1.f);
  float s = 0.f;
  for (int d = 0; d < 64; ++d) s += pool[g * 64 + d] * w[d * OUTD + o];
  out[i] = s / cc + b[o];
}

// ---------------- launch ----------------

extern "C" void kernel_launch(void* const* d_in, const int* in_sizes, int n_in,
                              void* d_out, int out_size, void* d_ws, size_t ws_size,
                              hipStream_t stream) {
  const float* x      = (const float*)d_in[0];
  const float* ea     = (const float*)d_in[1];
  const int*   eidx   = (const int*)d_in[2];
  const int*   batch  = (const int*)d_in[3];
  const float* fc1_w  = (const float*)d_in[4];
  const float* fc1_b  = (const float*)d_in[5];
  const float* e1w1   = (const float*)d_in[6];
  const float* e1b1   = (const float*)d_in[7];
  const float* e1w2   = (const float*)d_in[8];
  const float* e1b2   = (const float*)d_in[9];
  const float* root1  = (const float*)d_in[10];
  const float* bias1  = (const float*)d_in[11];
  const float* gn1w   = (const float*)d_in[12];
  const float* gn1b   = (const float*)d_in[13];
  const float* e2w1   = (const float*)d_in[14];
  const float* e2b1   = (const float*)d_in[15];
  const float* e2w2   = (const float*)d_in[16];
  const float* e2b2   = (const float*)d_in[17];
  const float* root2  = (const float*)d_in[18];
  const float* bias2  = (const float*)d_in[19];
  const float* gn2w   = (const float*)d_in[20];
  const float* gn2b   = (const float*)d_in[21];
  const float* fc2w   = (const float*)d_in[22];
  const float* fc2b   = (const float*)d_in[23];
  const int* src = eidx;
  const int* dst = eidx + EE;

  // workspace layout (256B aligned slabs)
  size_t off = 0;
  auto take = [&](size_t bytes) { size_t r = off; off = (off + bytes + 255) & ~(size_t)255; return r; };
  char* ws = (char*)d_ws;
  unsigned short* hxbf  = (unsigned short*)(ws + take((size_t)NN * 64 * 2));
  float*          h1    = (float*)         (ws + take((size_t)NN * 64 * 4));
  unsigned short* h1bf  = (unsigned short*)(ws + take((size_t)NN * 64 * 2));
  float*          agg   = (float*)         (ws + take((size_t)NN * 64 * 4));
  float*          h2    = (float*)         (ws + take((size_t)NN * 64 * 4));
  unsigned*       Hdup  = (unsigned*)      (ws + take((size_t)EE * 64 * 4));
  unsigned short* eabf  = (unsigned short*)(ws + take((size_t)EE * 32 * 2));
  unsigned short* xbf   = (unsigned short*)(ws + take((size_t)NN * 32 * 2));
  unsigned short* w2p1  = (unsigned short*)(ws + take((size_t)W2KS * 4 * 32 * 16 * 2));
  unsigned short* w2p2  = (unsigned short*)(ws + take((size_t)W2KS * 4 * 32 * 16 * 2));
  unsigned short* fc1p  = (unsigned short*)(ws + take(2048 * 2));
  unsigned short* e1p   = (unsigned short*)(ws + take(2048 * 2));
  unsigned short* e2p   = (unsigned short*)(ws + take(2048 * 2));
  unsigned short* r1p   = (unsigned short*)(ws + take(4096 * 2));
  unsigned short* r2p   = (unsigned short*)(ws + take(4096 * 2));
  float*          misc  = (float*)         (ws + take((4 + GG * 64 + GG) * 4));
  // misc: [0..3] stats, [4..4099] pool, [4100..4163] cnt

  const int TB = 256;
  auto cdiv = [](int a, int b) { return (a + b - 1) / b; };
  const int Ntiles  = NN / 16;   // 2048
  const int Etiles  = EE / 16;   // 6144
  const int Ewtiles = EE / 32;   // 3072 (32 edges per wave in nnconv_msg)

  zero_f32<<<cdiv(4 + GG * 64 + GG, TB), TB, 0, stream>>>(misc, 4 + GG * 64 + GG);

  // weight packing (cheap, recomputed each call for determinism)
  pack_w<<<cdiv(2048, TB), TB, 0, stream>>>(fc1_w, 32, 1, fc1p);
  pack_w<<<cdiv(2048, TB), TB, 0, stream>>>(e1w1, 16, 1, e1p);
  pack_w<<<cdiv(2048, TB), TB, 0, stream>>>(e2w1, 16, 1, e2p);
  pack_w<<<cdiv(4096, TB), TB, 0, stream>>>(root1, 64, 2, r1p);
  pack_w<<<cdiv(4096, TB), TB, 0, stream>>>(root2, 64, 2, r2p);
  pack_w2<<<cdiv(W2KS * 2048, TB), TB, 0, stream>>>(e1w2, e1b2, w2p1);
  pack_w2<<<cdiv(W2KS * 2048, TB), TB, 0, stream>>>(e2w2, e2b2, w2p2);

  // input conversion to bf16 (edge_attr padded K=16 -> 32)
  cvt_bf16<<<cdiv(NN * 32, TB), TB, 0, stream>>>(x, xbf, NN, 32, 32);
  cvt_bf16<<<cdiv(EE * 32, TB), TB, 0, stream>>>(ea, eabf, EE, 16, 32);

  // hx = x @ fc1_w + fc1_b  (bf16 output only; f32 never needed downstream)
  gemm16<<<Ntiles / 4, 128, 0, stream>>>(xbf, 32, Ntiles, 1, fc1p, fc1_b,
                                         nullptr, hxbf, nullptr, 0);

  // ---- conv layer 1 ----
  gemm16<<<Etiles / 4, 128, 0, stream>>>(eabf, 32, Etiles, 1, e1p, e1b1,
                                         nullptr, nullptr, Hdup, 1);      // H1 (relu)
  gemm16<<<Ntiles / 4, 128, 0, stream>>>(hxbf, 64, Ntiles, 2, r1p, bias1,
                                         agg, nullptr, nullptr, 0);       // agg = hx@root1 + bias1
  nnconv_msg<<<Ewtiles / 4, 128, 0, stream>>>(hxbf, Hdup, w2p1, src, dst, agg, Ewtiles);
  stats_k<<<1024, 256, 0, stream>>>(agg, (size_t)NN * 64, misc + 0);
  ln_relu<<<cdiv(NN * 64, TB), TB, 0, stream>>>(agg, misc + 0, gn1w, gn1b,
                                                h1, h1bf, NN * 64);

  // ---- conv layer 2 ----
  gemm16<<<Etiles / 4, 128, 0, stream>>>(eabf, 32, Etiles, 1, e2p, e2b1,
                                         nullptr, nullptr, Hdup, 1);      // H2 (relu)
  gemm16<<<Ntiles / 4, 128, 0, stream>>>(h1bf, 64, Ntiles, 2, r2p, bias2,
                                         agg, nullptr, nullptr, 0);       // agg = h1@root2 + bias2
  nnconv_msg<<<Ewtiles / 4, 128, 0, stream>>>(h1bf, Hdup, w2p2, src, dst, agg, Ewtiles);
  stats_k<<<1024, 256, 0, stream>>>(agg, (size_t)NN * 64, misc + 2);
  ln_relu<<<cdiv(NN * 64, TB), TB, 0, stream>>>(agg, misc + 2, gn2w, gn2b,
                                                h2, nullptr, NN * 64);

  // ---- pooling + head ----
  pool_k<<<cdiv(NN * 64, TB), TB, 0, stream>>>(h2, batch, misc + 4, misc + 4 + GG * 64, NN);
  head_k<<<cdiv(GG * OUTD, TB), TB, 0, stream>>>(misc + 4, misc + 4 + GG * 64,
                                                 fc2w, fc2b, (float*)d_out);
}
